// CodeSearchEncoder_70205535420855
// MI455X (gfx1250) — matrix operand order repaired and checked by reading, not compiled, not run on hardware
//
#include <hip/hip_runtime.h>
#include <hip/hip_bf16.h>
#include <math.h>

// ---------------- problem constants (from reference) ----------------
#define BB   4
#define LL   512
#define VV   16384
#define DD   384
#define NLAY 8
#define DSN  16          // state size
#define DCV  4           // conv taps
#define DI   768         // D * EXP
#define DTR  24
#define NXP  56          // DTR + 2*DSN
#define NXPP 64          // padded to WMMA tile multiple
#define EDIM 256
#define NTOK (BB*LL)     // 2048

typedef _Float16 half_t;
typedef __attribute__((ext_vector_type(16))) _Float16 v16h;
typedef __attribute__((ext_vector_type(8)))  _Float16 v8h;
typedef __attribute__((ext_vector_type(8)))  float    v8f;

union V16U { v16h v; v8h h[2]; };

__device__ __forceinline__ float silu_f(float x) { return x / (1.0f + __expf(-x)); }
__device__ __forceinline__ float softplus_f(float x) {
  return (x > 20.0f) ? x : log1pf(__expf(x));
}

// ---------------- f32 -> f16 conversion ----------------
__global__ void f32_to_f16_kernel(const float* __restrict__ in, half_t* __restrict__ out, int n) {
  int i = blockIdx.x * blockDim.x + threadIdx.x;
  if (i < n) out[i] = (half_t)in[i];
}

// ---------------- x_proj weight: (NL,56,DI) f32 -> (NL,64,DI) f16 zero-padded ----------------
__global__ void xw_pad_f16_kernel(const float* __restrict__ w, half_t* __restrict__ out) {
  int idx = blockIdx.x * blockDim.x + threadIdx.x;      // over NLAY*NXPP*DI
  if (idx >= NLAY * NXPP * DI) return;
  int e = idx % DI;
  int t = idx / DI;
  int o = t % NXPP;
  int layer = t / NXPP;
  out[idx] = (o < NXP) ? (half_t)w[((size_t)layer * NXP + o) * DI + e] : (half_t)0.0f;
}

// ---------------- embedding + positional ----------------
__global__ void embed_kernel(const int* __restrict__ ids, const float* __restrict__ emb,
                             const float* __restrict__ pos, float* __restrict__ x) {
  int idx = blockIdx.x * blockDim.x + threadIdx.x;      // over NTOK*DD
  if (idx >= NTOK * DD) return;
  int d = idx % DD;
  int t = idx / DD;
  int l = t % LL;
  int tok = ids[t];
  x[idx] = emb[(size_t)tok * DD + d] + pos[(size_t)l * DD + d];
}

// ---------------- layernorm (D=384), block=128, one row per block ----------------
__global__ __launch_bounds__(128)
void layernorm_kernel(const float* __restrict__ x, const float* __restrict__ g,
                      const float* __restrict__ b, half_t* __restrict__ out_h,
                      float* __restrict__ out_f) {
  int row = blockIdx.x;
  int tid = threadIdx.x;
  const float* xr = x + (size_t)row * DD;
  float v0 = xr[tid], v1 = xr[tid + 128], v2 = xr[tid + 256];
  __shared__ float red[128];
  red[tid] = v0 + v1 + v2;
  __syncthreads();
  for (int off = 64; off > 0; off >>= 1) {
    if (tid < off) red[tid] += red[tid + off];
    __syncthreads();
  }
  float mean = red[0] * (1.0f / DD);
  __syncthreads();
  float d0 = v0 - mean, d1 = v1 - mean, d2 = v2 - mean;
  red[tid] = d0 * d0 + d1 * d1 + d2 * d2;
  __syncthreads();
  for (int off = 64; off > 0; off >>= 1) {
    if (tid < off) red[tid] += red[tid + off];
    __syncthreads();
  }
  float rstd = rsqrtf(red[0] * (1.0f / DD) + 1e-5f);
  float dv[3] = {d0, d1, d2};
  for (int j = 0; j < 3; ++j) {
    int c = tid + j * 128;
    float val = dv[j] * rstd * g[c] + b[c];
    if (out_h) out_h[(size_t)row * DD + c] = (half_t)val;
    if (out_f) out_f[(size_t)row * DD + c] = val;
  }
}

// ---------------- WMMA f16 GEMM: C[M,N] (+)= A[M,K] * W[N,K]^T ----------------
// One wave -> one 16x64 strip of C (4 N-tiles, A fragment reused across them).
// All 5 fragments (A + 4xB) are issued before the WMMA chain so loads overlap
// with matrix math via staggered s_wait_loadcnt.
// Requires M mult of 16, N mult of 64, K mult of 32.
__global__ __launch_bounds__(256)
void gemm_wmma_f16(const half_t* __restrict__ A, const half_t* __restrict__ W,
                   float* __restrict__ C, int M, int N, int K, int accumulate) {
  int wave  = (blockIdx.x * blockDim.x + threadIdx.x) >> 5;
  int lane  = threadIdx.x & 31;
  int ngrp  = N >> 6;                      // groups of 64 columns
  int tileM = wave / ngrp;
  int tg    = wave - tileM * ngrp;
  if (tileM * 16 >= M) return;             // wave-uniform exit (EXEC stays all-1s)

  int hs = lane >> 4;                      // half-wave select
  int mr = lane & 15;
  const half_t* arow = A + (size_t)(tileM * 16 + mr) * K;
  int aoff = hs ? 8 : 0;                   // A: lanes16-31 start at K+8
  int boff = hs ? 8 : 0;                   // B: lanes16-31 start at K+8 (same as A)

  const half_t* wrow[4];
  #pragma unroll
  for (int j = 0; j < 4; ++j)
    wrow[j] = W + (size_t)(tg * 64 + j * 16 + mr) * K;

  v8f acc[4];
  #pragma unroll
  for (int j = 0; j < 4; ++j) acc[j] = (v8f){};

  for (int k0 = 0; k0 < K; k0 += 32) {
    if (k0 + 32 < K) {                     // prefetch next K slab (global_prefetch_b8)
      __builtin_prefetch(arow + k0 + 32, 0, 3);
      #pragma unroll
      for (int j = 0; j < 4; ++j) __builtin_prefetch(wrow[j] + k0 + 32, 0, 3);
    }
    // ---- stage all fragments first: 10 b128 loads in flight ----
    V16U au;
    au.h[0] = *(const v8h*)(arow + k0 + aoff);
    au.h[1] = *(const v8h*)(arow + k0 + aoff + 16);
    V16U bu[4];
    #pragma unroll
    for (int j = 0; j < 4; ++j) {
      bu[j].h[0] = *(const v8h*)(wrow[j] + k0 + boff);
      bu[j].h[1] = *(const v8h*)(wrow[j] + k0 + boff + 16);
    }
    // ---- WMMA chain: A reused, B/acc independent per step ----
    #pragma unroll
    for (int j = 0; j < 4; ++j) {
      acc[j] = __builtin_amdgcn_wmma_f32_16x16x32_f16(
          /*neg_a=*/false, au.v, /*neg_b=*/false, bu[j].v,
          /*c_mod=*/(short)0, acc[j], /*reuse_a=*/false, /*reuse_b=*/false);
    }
  }
  // C layout: lane holds column n; VGPR r holds row tileM*16 + r + 8*hs
  int mbase = tileM * 16 + hs * 8;
  #pragma unroll
  for (int j = 0; j < 4; ++j) {
    int n = tg * 64 + j * 16 + mr;
    #pragma unroll
    for (int r = 0; r < 8; ++r) {
      size_t idx = (size_t)(mbase + r) * N + n;
      if (accumulate) C[idx] += acc[j][r];
      else            C[idx]  = acc[j][r];
    }
  }
}

// ---------------- causal conv (DC=4) + SiLU ----------------
__global__ void conv_silu_kernel(const float* __restrict__ xz, const float* __restrict__ cw,
                                 const float* __restrict__ cb, float* __restrict__ xconv,
                                 half_t* __restrict__ xconv_h) {
  int idx = blockIdx.x * blockDim.x + threadIdx.x;      // over NTOK*DI
  if (idx >= NTOK * DI) return;
  int e = idx % DI;
  int t = idx / DI;
  int l = t % LL;
  int b = t / LL;
  float acc = cb[e];
  #pragma unroll
  for (int k = 0; k < DCV; ++k) {
    int ls = l + k - (DCV - 1);
    if (ls >= 0)
      acc += xz[(size_t)(b * LL + ls) * (2 * DI) + e] * cw[e * DCV + k];
  }
  float s = silu_f(acc);
  xconv[idx]   = s;
  xconv_h[idx] = (half_t)s;
}

// ---------------- selective scan: one thread per (b,d), state in VGPRs ----------------
__global__ __launch_bounds__(256)
void scan_kernel(const float* __restrict__ dbc, const float* __restrict__ xconv,
                 const float* __restrict__ xz, const float* __restrict__ dtw,
                 const float* __restrict__ dtb, const float* __restrict__ Alog,
                 const float* __restrict__ Dp, half_t* __restrict__ y_h) {
  int b = blockIdx.x / (DI / 256);
  int d = (blockIdx.x % (DI / 256)) * 256 + threadIdx.x;

  float w[DTR];
  #pragma unroll
  for (int r = 0; r < DTR; ++r) w[r] = dtw[(size_t)d * DTR + r];
  float A[DSN];
  #pragma unroll
  for (int n = 0; n < DSN; ++n) A[n] = -__expf(Alog[(size_t)d * DSN + n]);
  float h[DSN];
  #pragma unroll
  for (int n = 0; n < DSN; ++n) h[n] = 0.0f;
  float bias = dtb[d];
  float dval = Dp[d];

  __shared__ float s[NXPP];              // [0..23]=dt_r  [24..39]=B  [40..55]=C
  for (int l = 0; l < LL; ++l) {
    if (threadIdx.x < NXPP)
      s[threadIdx.x] = dbc[(size_t)(b * LL + l) * NXPP + threadIdx.x];
    __syncthreads();
    float dtv = bias;
    #pragma unroll
    for (int r = 0; r < DTR; ++r) dtv += s[r] * w[r];
    float dt = softplus_f(dtv);
    size_t row = (size_t)(b * LL + l);
    float xc = xconv[row * DI + d];
    float acc = 0.0f;
    #pragma unroll
    for (int n = 0; n < DSN; ++n) {
      h[n] = __expf(dt * A[n]) * h[n] + dt * s[24 + n] * xc;
      acc += h[n] * s[40 + n];
    }
    float zz = xz[row * (2 * DI) + DI + d];
    float yv = (acc + dval * xc) * silu_f(zz);
    y_h[row * DI + d] = (half_t)yv;
    __syncthreads();
  }
}

// ---------------- masked mean pool ----------------
__global__ __launch_bounds__(128)
void pool_kernel(const float* __restrict__ xf, const int* __restrict__ ids,
                 float* __restrict__ pooled) {
  int b = blockIdx.x;
  int d = blockIdx.y * 128 + threadIdx.x;
  float s = 0.0f, cnt = 0.0f;
  for (int l = 0; l < LL; ++l) {
    float m = (ids[b * LL + l] != 0) ? 1.0f : 0.0f;
    s   += xf[(size_t)(b * LL + l) * DD + d] * m;
    cnt += m;
  }
  pooled[(size_t)b * DD + d] = s / fmaxf(cnt, 1.0f);
}

// ---------------- final projection + L2 normalize ----------------
__global__ __launch_bounds__(256)
void proj_norm_kernel(const float* __restrict__ pooled, const float* __restrict__ pw,
                      const float* __restrict__ pb, float* __restrict__ out) {
  int b = blockIdx.x;
  int o = threadIdx.x;                 // 0..255
  const float* pr = pooled + (size_t)b * DD;
  const float* wr = pw + (size_t)o * DD;
  float acc = pb[o];
  for (int d = 0; d < DD; ++d) acc += pr[d] * wr[d];
  __shared__ float sq[256];
  sq[o] = acc * acc;
  __syncthreads();
  for (int off = 128; off > 0; off >>= 1) {
    if (o < off) sq[o] += sq[o + off];
    __syncthreads();
  }
  float nrm = fmaxf(sqrtf(sq[0]), 1e-12f);
  out[(size_t)b * EDIM + o] = acc / nrm;
}

// ---------------- host orchestration ----------------
static inline size_t align256(size_t x) { return (x + 255) & ~(size_t)255; }

extern "C" void kernel_launch(void* const* d_in, const int* in_sizes, int n_in,
                              void* d_out, int out_size, void* d_ws, size_t ws_size,
                              hipStream_t stream) {
  const int*   input_ids = (const int*)  d_in[0];
  const float* emb       = (const float*)d_in[1];
  const float* pos_emb   = (const float*)d_in[2];
  const float* ln_g      = (const float*)d_in[3];
  const float* ln_b      = (const float*)d_in[4];
  const float* in_proj_w = (const float*)d_in[5];
  const float* conv_w    = (const float*)d_in[6];
  const float* conv_b    = (const float*)d_in[7];
  const float* x_proj_w  = (const float*)d_in[8];
  const float* dt_proj_w = (const float*)d_in[9];
  const float* dt_proj_b = (const float*)d_in[10];
  const float* A_log     = (const float*)d_in[11];
  const float* D_param   = (const float*)d_in[12];
  const float* out_proj_w= (const float*)d_in[13];
  const float* fln_g     = (const float*)d_in[14];
  const float* fln_b     = (const float*)d_in[15];
  const float* proj_w    = (const float*)d_in[16];
  const float* proj_b    = (const float*)d_in[17];
  float* out = (float*)d_out;

  // ---- workspace carve-up ----
  char* ws = (char*)d_ws;
  size_t off = 0;
  #define ALLOC(ptr, type, count) type* ptr = (type*)(ws + off); off = align256(off + (size_t)(count) * sizeof(type))
  ALLOC(in_w_h,  half_t, (size_t)NLAY * 2 * DI * DD);   // 8*1536*384
  ALLOC(ow_h,    half_t, (size_t)NLAY * DD * DI);       // 8*384*768
  ALLOC(xw_h,    half_t, (size_t)NLAY * NXPP * DI);     // 8*64*768 zero-padded
  ALLOC(xres,    float,  (size_t)NTOK * DD);            // residual stream
  ALLOC(xn_h,    half_t, (size_t)NTOK * DD);
  ALLOC(xz,      float,  (size_t)NTOK * 2 * DI);
  ALLOC(xconv,   float,  (size_t)NTOK * DI);
  ALLOC(xconv_h, half_t, (size_t)NTOK * DI);
  ALLOC(dbc,     float,  (size_t)NTOK * NXPP);
  ALLOC(y_h,     half_t, (size_t)NTOK * DI);
  ALLOC(xf,      float,  (size_t)NTOK * DD);
  ALLOC(pooled,  float,  (size_t)BB * DD);
  #undef ALLOC
  (void)ws_size; (void)in_sizes; (void)n_in; (void)out_size;

  // ---- one-time weight conversion to f16 ----
  {
    int n1 = NLAY * 2 * DI * DD;
    f32_to_f16_kernel<<<(n1 + 255) / 256, 256, 0, stream>>>(in_proj_w, in_w_h, n1);
    int n2 = NLAY * DD * DI;
    f32_to_f16_kernel<<<(n2 + 255) / 256, 256, 0, stream>>>(out_proj_w, ow_h, n2);
    int n3 = NLAY * NXPP * DI;
    xw_pad_f16_kernel<<<(n3 + 255) / 256, 256, 0, stream>>>(x_proj_w, xw_h);
  }

  // ---- embedding ----
  embed_kernel<<<(NTOK * DD + 255) / 256, 256, 0, stream>>>(input_ids, emb, pos_emb, xres);

  // ---- layers ----
  for (int i = 0; i < NLAY; ++i) {
    layernorm_kernel<<<NTOK, 128, 0, stream>>>(
        xres, ln_g + i * DD, ln_b + i * DD, xn_h, (float*)nullptr);

    // in_proj: [2048,384] x [1536,384]^T -> [2048,1536]   (128 * 24 = 3072 waves)
    {
      int waves = (NTOK / 16) * ((2 * DI) / 64);
      gemm_wmma_f16<<<waves / 8, 256, 0, stream>>>(
          xn_h, in_w_h + (size_t)i * 2 * DI * DD, xz, NTOK, 2 * DI, DD, 0);
    }

    conv_silu_kernel<<<(NTOK * DI + 255) / 256, 256, 0, stream>>>(
        xz, conv_w + (size_t)i * DI * DCV, conv_b + (size_t)i * DI, xconv, xconv_h);

    // x_proj (padded N=64): [2048,768] x [64,768]^T -> [2048,64]   (128 waves)
    {
      int waves = (NTOK / 16) * (NXPP / 64);
      gemm_wmma_f16<<<waves / 8, 256, 0, stream>>>(
          xconv_h, xw_h + (size_t)i * NXPP * DI, dbc, NTOK, NXPP, DI, 0);
    }

    scan_kernel<<<BB * (DI / 256), 256, 0, stream>>>(
        dbc, xconv, xz,
        dt_proj_w + (size_t)i * DI * DTR, dt_proj_b + (size_t)i * DI,
        A_log + (size_t)i * DI * DSN, D_param + (size_t)i * DI, y_h);

    // out_proj fused with residual: xres += [2048,768] x [384,768]^T   (128*6 = 768 waves)
    {
      int waves = (NTOK / 16) * (DD / 64);
      gemm_wmma_f16<<<waves / 8, 256, 0, stream>>>(
          y_h, ow_h + (size_t)i * DD * DI, xres, NTOK, DD, DI, 1);
    }
  }

  // ---- final layernorm, pool, projection + normalize ----
  layernorm_kernel<<<NTOK, 128, 0, stream>>>(xres, fln_g, fln_b, (half_t*)nullptr, xf);
  pool_kernel<<<dim3(BB, DD / 128), 128, 0, stream>>>(xf, input_ids, pooled);
  proj_norm_kernel<<<BB, 256, 0, stream>>>(pooled, proj_w, proj_b, out);
}
